// MuSc_14096082666330
// MI455X (gfx1250) — compile-verified, matrix operand order checked by reading, not run on hardware
//
#include <hip/hip_runtime.h>

typedef __attribute__((ext_vector_type(16))) __bf16       v16bf;
typedef __attribute__((ext_vector_type(8)))  float        v8f;
typedef __attribute__((ext_vector_type(4)))  unsigned int v4u;
typedef __attribute__((ext_vector_type(8)))  int          v8i;
typedef __attribute__((ext_vector_type(4)))  int          v4i;

#define BB   8
#define PHW  16     // patches per side
#define PP   256    // patches per image
#define NN   2048   // BB*PP
#define DD   1024
#define LL   4
#define CC   588    // 3*14*14
#define HH   224
#define KTOP 2
#define NB1  7      // BB-1

// ---------------------------------------------------------------------------
// 0) zero the score accumulator
__global__ void musc_zero(float* __restrict__ sc) {
  int i = blockIdx.x * 256 + threadIdx.x;
  if (i < NN) sc[i] = 0.f;
}

// ---------------------------------------------------------------------------
// 1) patch extraction + linear embedding (f32): feats[l][i][d]
__global__ __launch_bounds__(256)
void musc_embed(const float* __restrict__ pix, const float* __restrict__ Wv,
                const float* __restrict__ bv, float* __restrict__ feats) {
  __shared__ float At[16 * CC];                 // 37.6 KB
  const int l  = blockIdx.z;
  const int rt = blockIdx.y;                    // row tile (16 rows)
  const int cb = blockIdx.x;                    // col block (128 cols)
  const int tid = threadIdx.x;

  for (int t = tid; t < 16 * CC; t += 256) {
    int rr = t / CC, c = t - rr * CC;
    int gi = rt * 16 + rr;
    int b = gi >> 8, p = gi & 255;
    int py = p >> 4, px = p & 15;
    int ch = c / 196, rem = c - ch * 196;
    int dy = rem / 14, dx = rem - dy * 14;
    At[t] = pix[((size_t)(b * 3 + ch) * HH + (py * 14 + dy)) * HH + (px * 14 + dx)];
  }
  __syncthreads();

  const int col = cb * 128 + (tid & 127);
  const int rb  = (tid >> 7) * 8;               // rows rb..rb+7
  float acc[8];
#pragma unroll
  for (int j = 0; j < 8; ++j) acc[j] = 0.f;
  const float* wp = Wv + (size_t)l * CC * DD + col;
  for (int k = 0; k < CC; ++k) {
    float w = wp[(size_t)k * DD];
#pragma unroll
    for (int j = 0; j < 8; ++j)
      acc[j] = fmaf(At[(rb + j) * CC + k], w, acc[j]);
  }
  const float bias = bv[l * DD + col];
#pragma unroll
  for (int j = 0; j < 8; ++j) {
    int gi = rt * 16 + rb + j;
    feats[((size_t)l * NN + gi) * DD + col] = acc[j] + bias;
  }
}

// ---------------------------------------------------------------------------
// 2) LayerNorm in place over D=1024 (one block per row). r-independent.
__global__ __launch_bounds__(256)
void musc_ln(float* __restrict__ f) {
  __shared__ float s1[256], s2[256];
  const int tid = threadIdx.x;
  float* x = f + (size_t)blockIdx.x * DD;
  float v[4], sum = 0.f, sq = 0.f;
#pragma unroll
  for (int k = 0; k < 4; ++k) {
    v[k] = x[tid + k * 256];
    sum += v[k]; sq += v[k] * v[k];
  }
  s1[tid] = sum; s2[tid] = sq; __syncthreads();
  for (int off = 128; off > 0; off >>= 1) {
    if (tid < off) { s1[tid] += s1[tid + off]; s2[tid] += s2[tid + off]; }
    __syncthreads();
  }
  const float mu  = s1[0] * (1.f / DD);
  const float var = s2[0] * (1.f / DD) - mu * mu;
  const float rs  = rsqrtf(var + 1e-5f);
#pragma unroll
  for (int k = 0; k < 4; ++k) x[tid + k * 256] = (v[k] - mu) * rs;
}

// ---------------------------------------------------------------------------
// 3) r x r zero-padded neighborhood mean + L2 normalize -> bf16 rf
__global__ __launch_bounds__(256)
void musc_pool(const float* __restrict__ ln, __bf16* __restrict__ rf, int r) {
  __shared__ float sred[256];
  const int tid = threadIdx.x;
  const int flat = blockIdx.x;
  const int l = flat >> 11, i = flat & 2047;
  const int b = i >> 8, p = i & 255, py = p >> 4, px = p & 15;
  const int h = r >> 1;
  const float inv = 1.f / (float)(r * r);
  const float* base = ln + ((size_t)l * NN + b * PP) * DD;

  float vals[4], ss = 0.f;
#pragma unroll
  for (int k = 0; k < 4; ++k) {
    const int d = tid + k * 256;
    float s = 0.f;
    for (int dy = -h; dy <= h; ++dy) {
      const int yy = py + dy;
      if (yy < 0 || yy >= PHW) continue;
      for (int dx = -h; dx <= h; ++dx) {
        const int xx = px + dx;
        if (xx < 0 || xx >= PHW) continue;
        s += base[(size_t)(yy * PHW + xx) * DD + d];
      }
    }
    vals[k] = s * inv;
    ss += vals[k] * vals[k];
  }
  sred[tid] = ss; __syncthreads();
  for (int off = 128; off > 0; off >>= 1) {
    if (tid < off) sred[tid] += sred[tid + off];
    __syncthreads();
  }
  const float rn = rsqrtf(fmaxf(sred[0], 1e-24f));
  __bf16* out = rf + ((size_t)l * NN + i) * DD;
#pragma unroll
  for (int k = 0; k < 4; ++k) out[tid + k * 256] = (__bf16)(vals[k] * rn);
}

// ---------------------------------------------------------------------------
// 4) WMMA Gram + fused min/argmin over 256 ref patches of one target batch.
//    A-tile (16x1024 bf16, 32 KB contiguous) staged via the Tensor Data Mover.
//    Reduction runs on raw dot products (argmin dist == argmax dot); a single
//    sqrt per final output. Each wave drives two independent WMMA chains whose
//    accumulators are merged element-wise before one butterfly per pair.
//    grid = (targetBatch=8, rowTile=128, layer=4), block = 128 (4 waves).
__global__ __launch_bounds__(128)
void musc_grammin(const __bf16* __restrict__ rf, float* __restrict__ md,
                  int* __restrict__ am) {
  __shared__ __bf16 At[16 * DD];                // 32 KB A-row block
  __shared__ float  redv[4][16];
  __shared__ int    redi[4][16];
  const int tb = blockIdx.x, rt = blockIdx.y, l = blockIdx.z;
  const int tid = threadIdx.x;
  const int wave = tid >> 5, lane = tid & 31;
  const int hs = lane >> 4, mrow = lane & 15;   // K-half select, row/col-in-tile

  const __bf16* Ag = rf + ((size_t)l * NN + rt * 16) * DD;
#if __has_builtin(__builtin_amdgcn_tensor_load_to_lds)
  if (wave == 0) {
    // 1-D TDM descriptor: 4096 x 8-byte elements, contiguous (ISA 8.3/8.4)
    unsigned long long ga = (unsigned long long)(uintptr_t)Ag;
    unsigned lds = (unsigned)(uintptr_t)&At[0];
    v4u g0 = { 1u,                                        // count=1, user mode
               lds,                                       // lds_addr (bytes)
               (unsigned)(ga & 0xffffffffu),              // global_addr[31:0]
               (unsigned)((ga >> 32) & 0x01ffffffu)       // global_addr[56:32]
                   | 0x80000000u };                       // type=2 ("image")
    v8i g1 = { (int)(3u << 16),       // data_size=8B, mask=0 (not in cluster)
               (int)(4096u << 16),    // tensor_dim0[15:0]=4096 at bits[63:48]
               (int)(1u << 16),       // tensor_dim1=1 at bits[95:80]
               (int)(4096u << 16),    // tile_dim0=4096 at bits[127:112]
               1,                     // tile_dim1=1
               4096,                  // tensor_dim0_stride=4096
               0, 0 };
    v4i gz4 = {0, 0, 0, 0};
    v8i gz8 = {0, 0, 0, 0, 0, 0, 0, 0};
    __builtin_amdgcn_tensor_load_to_lds(g0, g1, gz4, gz4, gz8, 0);
    __builtin_amdgcn_s_wait_tensorcnt(0);
  }
#else
  {
    const v4u* src = (const v4u*)Ag;
    v4u* dst = (v4u*)At;
    for (int v = tid; v < 16 * DD / 8; v += 128) dst[v] = src[v];
  }
#endif
  __syncthreads();

  float rmax[8]; int ridx[8];                   // running max dot + argmax
#pragma unroll
  for (int j = 0; j < 8; ++j) { rmax[j] = -3.4e38f; ridx[j] = 0; }

  const size_t browbase = ((size_t)l * NN + tb * PP) * DD;
#pragma unroll 1
  for (int pi = 0; pi < 2; ++pi) {              // 2 pairs of column tiles
    const int ct0 = wave * 4 + pi * 2;
    const int ct1 = ct0 + 1;
    v8f acc0 = {0.f, 0.f, 0.f, 0.f, 0.f, 0.f, 0.f, 0.f};
    v8f acc1 = {0.f, 0.f, 0.f, 0.f, 0.f, 0.f, 0.f, 0.f};
    const __bf16* B0 = rf + browbase + (size_t)(ct0 * 16 + mrow) * DD;
    const __bf16* B1 = rf + browbase + (size_t)(ct1 * 16 + mrow) * DD;
#pragma unroll 2
    for (int kt = 0; kt < DD / 32; ++kt) {
      const int k0 = kt * 32 + hs * 8;          // lane K layout (ISA 7.12.2)
      union { v4u u[2]; v16bf v; } a, b0u, b1u;
      a.u[0]   = *(const v4u*)&At[mrow * DD + k0];
      a.u[1]   = *(const v4u*)&At[mrow * DD + k0 + 16];
      b0u.u[0] = *(const v4u*)&B0[k0];
      b0u.u[1] = *(const v4u*)&B0[k0 + 16];
      b1u.u[0] = *(const v4u*)&B1[k0];
      b1u.u[1] = *(const v4u*)&B1[k0 + 16];
      acc0 = __builtin_amdgcn_wmma_f32_16x16x32_bf16(
          false, a.v, false, b0u.v, (short)0, acc0, false, false);
      acc1 = __builtin_amdgcn_wmma_f32_16x16x32_bf16(
          false, a.v, false, b1u.v, (short)0, acc1, false, false);
    }
    // merge the two tiles element-wise, then one butterfly over 16 columns
    float dj[8]; int ij[8];
#pragma unroll
    for (int j = 0; j < 8; ++j) {
      const float a0 = acc0[j], a1 = acc1[j];
      const bool t = a1 > a0;
      dj[j] = t ? a1 : a0;
      ij[j] = (t ? ct1 : ct0) * 16 + mrow;
    }
#pragma unroll
    for (int mask = 1; mask <= 8; mask <<= 1) {
#pragma unroll
      for (int j = 0; j < 8; ++j) {
        float od = __shfl_xor(dj[j], mask, 32);
        int   oi = __shfl_xor(ij[j], mask, 32);
        if (od > dj[j]) { dj[j] = od; ij[j] = oi; }
      }
    }
#pragma unroll
    for (int j = 0; j < 8; ++j)
      if (dj[j] > rmax[j]) { rmax[j] = dj[j]; ridx[j] = ij[j]; }
  }

  if (mrow == 0) {
#pragma unroll
    for (int j = 0; j < 8; ++j) {
      redv[wave][hs * 8 + j] = rmax[j];
      redi[wave][hs * 8 + j] = ridx[j];
    }
  }
  __syncthreads();
  if (tid < 16) {
    float bg = redv[0][tid]; int bi = redi[0][tid];
#pragma unroll
    for (int w = 1; w < 4; ++w)
      if (redv[w][tid] > bg) { bg = redv[w][tid]; bi = redi[w][tid]; }
    const int gi = l * NN + rt * 16 + tid;
    md[(size_t)gi * BB + tb] = sqrtf(fmaxf(2.f - 2.f * bg, 1e-12f));
    am[(size_t)gi * BB + tb] = bi;
  }
}

// ---------------------------------------------------------------------------
// 5) exclude self, top-2 smallest of 7, write outputs, accumulate scores
__global__ __launch_bounds__(256)
void musc_score(const float* __restrict__ md, const int* __restrict__ am,
                float* __restrict__ out_min, float* __restrict__ out_tki,
                float* __restrict__ out_tks, float* __restrict__ scores,
                int ridx) {
  const int idx = blockIdx.x * 256 + threadIdx.x;
  if (idx >= NN) return;
  const int b = idx >> 8, p = idx & 255;
  float ssum = 0.f;
  for (int l = 0; l < LL; ++l) {
    const size_t base = ((size_t)l * NN + idx) * BB;
    float dv[NB1]; int av_[NB1];
    int j = 0;
    for (int tb = 0; tb < BB; ++tb) {
      if (tb == b) continue;
      dv[j] = md[base + tb];
      av_[j] = am[base + tb];
      ++j;
    }
    int i1 = 0;
    for (int t = 1; t < NB1; ++t) if (dv[t] < dv[i1]) i1 = t;
    int i2 = -1;
    for (int t = 0; t < NB1; ++t) {
      if (t == i1) continue;
      if (i2 < 0 || dv[t] < dv[i2]) i2 = t;
    }
    const size_t ob = ((size_t)b * LL + l) * 3 + ridx;
    for (int t = 0; t < NB1; ++t)
      out_min[(ob * NB1 + t) * PP + p] = (float)av_[t];
    out_tki[(ob * KTOP + 0) * PP + p] = (float)i1;
    out_tki[(ob * KTOP + 1) * PP + p] = (float)i2;
    out_tks[(ob * KTOP + 0) * PP + p] = dv[i1];
    out_tks[(ob * KTOP + 1) * PP + p] = dv[i2];
    ssum += 0.5f * (dv[i1] + dv[i2]);
  }
  scores[idx] += ssum * (1.f / (LL * 3));   // mean over layers and r
}

// ---------------------------------------------------------------------------
// 6) per-image max + argmax
__global__ __launch_bounds__(256)
void musc_final(const float* __restrict__ scores, float* __restrict__ fs,
                float* __restrict__ mi) {
  __shared__ float sv[256];
  __shared__ int   si[256];
  const int b = blockIdx.x, t = threadIdx.x;
  sv[t] = scores[b * PP + t]; si[t] = t; __syncthreads();
  for (int off = 128; off > 0; off >>= 1) {
    if (t < off) {
      if (sv[t + off] > sv[t] ||
          (sv[t + off] == sv[t] && si[t + off] < si[t])) {
        sv[t] = sv[t + off]; si[t] = si[t + off];
      }
    }
    __syncthreads();
  }
  if (t == 0) { fs[b] = sv[0]; mi[b] = (float)si[0]; }
}

// ---------------------------------------------------------------------------
// 7) align_corners=True bilinear 16x16 -> 224x224
__global__ __launch_bounds__(256)
void musc_upsample(const float* __restrict__ scores, float* __restrict__ out) {
  const int t = blockIdx.x * 256 + threadIdx.x;
  if (t >= BB * HH * HH) return;
  const int b = t / (HH * HH);
  const int rem = t - b * HH * HH;
  const int oy = rem / HH, ox = rem - oy * HH;
  const float ys = oy * (float)(PHW - 1) / (float)(HH - 1);
  const float xs = ox * (float)(PHW - 1) / (float)(HH - 1);
  int y0 = (int)floorf(ys), x0 = (int)floorf(xs);
  int y1 = min(y0 + 1, PHW - 1), x1 = min(x0 + 1, PHW - 1);
  const float wy = ys - y0, wx = xs - x0;
  const float* sb = scores + b * PP;
  const float Ia = sb[y0 * PHW + x0], Ib = sb[y0 * PHW + x1];
  const float Ic = sb[y1 * PHW + x0], Id = sb[y1 * PHW + x1];
  out[t] = Ia * (1 - wy) * (1 - wx) + Ib * (1 - wy) * wx +
           Ic * wy * (1 - wx) + Id * wy * wx;
}

// ---------------------------------------------------------------------------
extern "C" void kernel_launch(void* const* d_in, const int* in_sizes, int n_in,
                              void* d_out, int out_size, void* d_ws,
                              size_t ws_size, hipStream_t stream) {
  const float* pix = (const float*)d_in[0];   // [8,3,224,224]
  const float* Wv  = (const float*)d_in[1];   // [4,588,1024]
  const float* bvp = (const float*)d_in[2];   // [4,1024]

  char* ws = (char*)d_ws;
  float*  ln = (float*)ws;                              // 32 MB (feats -> LN in place)
  __bf16* rf = (__bf16*)(ws + (size_t)33554432);        // 16 MB, reused per r
  float*  md = (float*) (ws + (size_t)50331648);        // 256 KB
  int*    am = (int*)   (ws + (size_t)50593792);        // 256 KB
  float*  sc = (float*) (ws + (size_t)50855936);        // 8 KB

  float* o_final = (float*)d_out;                       // [8]
  float* o_pixel = o_final + BB;                        // [8,224,224]
  float* o_min   = o_pixel + (size_t)BB * HH * HH;      // [8,4,3,7,256]
  float* o_maxi  = o_min + (size_t)BB * LL * 3 * NB1 * PP;  // [8]
  float* o_tki   = o_maxi + BB;                         // [8,4,3,2,256]
  float* o_tks   = o_tki + (size_t)BB * LL * 3 * KTOP * PP; // [8,4,3,2,256]

  musc_zero<<<8, 256, 0, stream>>>(sc);
  musc_embed<<<dim3(8, 128, LL), 256, 0, stream>>>(pix, Wv, bvp, ln);
  musc_ln<<<LL * NN, 256, 0, stream>>>(ln);

  const int rlist[3] = {1, 3, 5};
  for (int ri = 0; ri < 3; ++ri) {
    musc_pool<<<LL * NN, 256, 0, stream>>>(ln, rf, rlist[ri]);
    musc_grammin<<<dim3(BB, NN / 16, LL), 128, 0, stream>>>(rf, md, am);
    musc_score<<<NN / 256, 256, 0, stream>>>(md, am, o_min, o_tki, o_tks, sc, ri);
  }

  musc_final<<<BB, 256, 0, stream>>>(sc, o_final, o_maxi);
  musc_upsample<<<(BB * HH * HH + 255) / 256, 256, 0, stream>>>(sc, o_pixel);
}
